// HyperLSTMCell_42649025249810
// MI455X (gfx1250) — matrix operand order repaired and checked by reading, not compiled, not run on hardware
//
#include <hip/hip_runtime.h>
#include <cmath>

#define B_N   16384
#define OBS_N 128
#define HL_N  256
#define H_N   256
#define HH_N  128
#define E_N   32
#define NC_N  512
#define TM    32      // rows per block tile
#define NTHR  256     // 8 waves (wave32)
#define EPSF  1e-6f

typedef __attribute__((ext_vector_type(16))) _Float16 v16h;
typedef __attribute__((ext_vector_type(8)))  _Float16 v8h;
typedef __attribute__((ext_vector_type(8)))  float    v8f;

union V16U { v16h v; v8h h[2]; };

// Output tensor offsets in d_out (floats), in reference return order:
// (output, new_h, new_c, new_hyper_h, new_hyper_c)
constexpr size_t OFF_OUT = 0;
constexpr size_t OFF_NH  = (size_t)B_N * NC_N;
constexpr size_t OFF_NC  = OFF_NH + (size_t)B_N * H_N;
constexpr size_t OFF_HH  = OFF_NC + (size_t)B_N * H_N;
constexpr size_t OFF_HC  = OFF_HH + (size_t)B_N * HH_N;

// ---------------- WMMA fragment helpers (CDNA5 16x16x32 f16, wave32) ----------

// A: 16x32 f16 tile, row-major in LDS with leading dim lda (halves).
// Lanes 0-15 hold row M=lane, K=0..15;
// lanes 16-31 hold row M=lane-16, K=16..31 (matches B fragment).
__device__ __forceinline__ v16h load_a_frag(const _Float16* A, int lda) {
  const int lane = threadIdx.x & 31;
  const int m  = lane & 15;
  const int kb = (lane >> 4) << 4;           // 0 or 16 halves
  V16U u;
  u.h[0] = *(const v8h*)(A + m * lda + kb);        // K = kb .. kb+7
  u.h[1] = *(const v8h*)(A + m * lda + kb + 8);    // K = kb+8 .. kb+15
  return u.v;
}

// B: 32x16 f16 tile = rows n of W (N x K row-major).  Same k order as A: lanes 0-15
// hold N=lane with K=0..15, lanes 16-31 hold N=lane-16 with K=16..31 -> one
// contiguous 32-byte load per lane.
__device__ __forceinline__ v16h load_b_frag(const _Float16* W, int ldb) {
  const int lane = threadIdx.x & 31;
  const int n  = lane & 15;
  const int kb = (lane >> 4) << 4;           // 0 or 16 halves
  return *(const v16h*)(W + (size_t)n * ldb + kb);
}

// Block-level GEMM: C(32 x N, f32 in LDS) = A(32 x K, f16 in LDS) * W^T,
// W is (N x K) row-major f16 in global (L2-resident).  8 waves arranged as
// 2 (M) x 4 (N) grid of 16x16 WMMA tiles, looping N in steps of 64.
template<int K, bool ACC>
__device__ void gemm_tile(const _Float16* __restrict__ A, int lda,
                          const _Float16* __restrict__ W, int N,
                          float* __restrict__ C, int ldc)
{
  const int wave = threadIdx.x >> 5;
  const int lane = threadIdx.x & 31;
  const int mrow = (wave >> 2) << 4;         // 0 or 16
  const int nn   = lane & 15;
  const int mo   = (lane >> 4) << 3;         // C frag row offset 0 or 8
  for (int n0 = (wave & 3) << 4; n0 < N; n0 += 64) {
    __builtin_prefetch(W + (size_t)(n0 + 64) * K, 0, 1);  // global_prefetch_b8
    v8f acc = {};
    if (ACC) {
#pragma unroll
      for (int r = 0; r < 8; ++r)
        acc[r] = C[(mrow + mo + r) * ldc + n0 + nn];
    }
#pragma unroll
    for (int k0 = 0; k0 < K; k0 += 32) {
      v16h a = load_a_frag(A + mrow * lda + k0, lda);
      v16h b = load_b_frag(W + (size_t)n0 * K + k0, K);
      acc = __builtin_amdgcn_wmma_f32_16x16x32_f16(
          /*neg_a=*/false, a, /*neg_b=*/false, b,
          /*c_mod=*/(short)0, acc, /*reuse_a=*/false, /*reuse_b=*/false);
    }
#pragma unroll
    for (int r = 0; r < 8; ++r)
      C[(mrow + mo + r) * ldc + n0 + nn] = acc[r];
  }
}

// Wave-per-row layernorm statistics over 256 contiguous floats in LDS.
// ddof=1 to match jnp.std(..., ddof=1); istd = 1/(std + eps).
__device__ __forceinline__ void row_stats(const float* __restrict__ row,
                                          float& mean, float& istd) {
  const int lane = threadIdx.x & 31;
  float s = 0.f, ss = 0.f;
#pragma unroll
  for (int q = 0; q < 8; ++q) {
    float v = row[lane + (q << 5)];
    s += v; ss += v * v;
  }
#pragma unroll
  for (int m = 16; m > 0; m >>= 1) {
    s  += __shfl_xor(s,  m, 32);
    ss += __shfl_xor(ss, m, 32);
  }
  mean = s * (1.f / 256.f);
  float var = (ss - 256.f * mean * mean) * (1.f / 255.f);
  istd = 1.f / (sqrtf(fmaxf(var, 0.f)) + EPSF);
}

__device__ __forceinline__ float sigf(float x) { return 1.f / (1.f + expf(-x)); }

// -------------------------- weight conversion kernel --------------------------

__global__ void f32_to_f16_kernel(const float* __restrict__ s,
                                  _Float16* __restrict__ d, int n) {
  int i = blockIdx.x * blockDim.x + threadIdx.x;
  if (i < n) d[i] = (_Float16)s[i];
}

// ------------------------------ fused cell kernel -----------------------------

__global__ __launch_bounds__(NTHR)
void hyperlstm_fused(
    const float* __restrict__ x,        const float* __restrict__ h,
    const float* __restrict__ c,        const float* __restrict__ hyper_h,
    const float* __restrict__ hyper_c,
    const float* __restrict__ enc_b1,   const float* __restrict__ enc_b2,
    const float* __restrict__ enc_b3,
    const float* __restrict__ bias,     const float* __restrict__ proj_b,
    const float* __restrict__ gamma_ifgo, const float* __restrict__ beta_ifgo,
    const float* __restrict__ gamma_c,  const float* __restrict__ beta_c,
    const float* __restrict__ out_b,    const float* __restrict__ hb,
    const _Float16* __restrict__ wEnc1, const _Float16* __restrict__ wEnc2,
    const _Float16* __restrict__ wEnc3, const _Float16* __restrict__ wIh,
    const _Float16* __restrict__ wHh,   const _Float16* __restrict__ wProj,
    const _Float16* __restrict__ wScale,const _Float16* __restrict__ wOut,
    const _Float16* __restrict__ wHIh,  const _Float16* __restrict__ wHHh,
    float* __restrict__ o)
{
  extern __shared__ unsigned char smem[];
  // ---- LDS partition (264 KB total, < 320 KB/WGP) ----
  _Float16* xh  = (_Float16*)smem;          // TM x 512  : [xe | h] (hyper_in)
  _Float16* xin = xh  + TM * 512;           // TM x 128  : encoder input
  _Float16* t1  = xin + TM * 128;           // TM x 256  : enc temp / new_h f16
  _Float16* t2  = t1  + TM * 256;           // TM x 256  : enc temp
  _Float16* hhb = t2  + TM * 256;           // TM x 128  : hyper_h -> new_hyper_h
  _Float16* zb  = hhb + TM * 128;           // TM x 384  : z (12 groups of 32)
  _Float16* act = zb  + TM * 384;           // 4 x TM x 256 : activated gates
  float*    Cb  = (float*)(act + 4 * TM * 256);  // TM x 512 f32 GEMM scratch
  float*    hB  = Cb + TM * 512;            // TM x 256 f32 extra scratch
  float*    gb  = Cb;                       // alias: TM x 256 (gate accum)
  float*    hA  = Cb + TM * 256;            // alias: TM x 256 (hv scratch)

  const int row0 = blockIdx.x * TM;
  const int tid  = threadIdx.x;
  const int wave = tid >> 5;
  const int lane = tid & 31;

  // ---- load input tiles, convert to f16 ----
  for (int i = tid; i < TM * OBS_N; i += NTHR)
    xin[i] = (_Float16)x[(size_t)(row0 + i / OBS_N) * OBS_N + (i % OBS_N)];
  for (int i = tid; i < TM * H_N; i += NTHR)
    xh[(i / H_N) * 512 + 256 + (i % H_N)] =
        (_Float16)h[(size_t)(row0 + i / H_N) * H_N + (i % H_N)];
  for (int i = tid; i < TM * HH_N; i += NTHR)
    hhb[i] = (_Float16)hyper_h[(size_t)(row0 + i / HH_N) * HH_N + (i % HH_N)];
  __syncthreads();

  // ---- encoder MLP: relu(W1 x), relu(W2 .), W3 . ----
  gemm_tile<OBS_N, false>(xin, OBS_N, wEnc1, HL_N, Cb, HL_N);
  __syncthreads();
  for (int i = tid; i < TM * HL_N; i += NTHR)
    t1[i] = (_Float16)fmaxf(Cb[i] + enc_b1[i % HL_N], 0.f);
  __syncthreads();
  gemm_tile<HL_N, false>(t1, HL_N, wEnc2, HL_N, Cb, HL_N);
  __syncthreads();
  for (int i = tid; i < TM * HL_N; i += NTHR)
    t2[i] = (_Float16)fmaxf(Cb[i] + enc_b2[i % HL_N], 0.f);
  __syncthreads();
  gemm_tile<HL_N, false>(t2, HL_N, wEnc3, HL_N, Cb, HL_N);
  __syncthreads();
  for (int i = tid; i < TM * HL_N; i += NTHR)
    xh[(i / HL_N) * 512 + (i % HL_N)] = (_Float16)(Cb[i] + enc_b3[i % HL_N]);
  __syncthreads();

  // ---- hyper gates: hg = [xe,h] @ hW_ih^T + hyper_h @ hW_hh^T ----
  gemm_tile<512, false>(xh, 512, wHIh, 4 * HH_N, Cb, 4 * HH_N);
  __syncthreads();
  gemm_tile<HH_N, true>(hhb, HH_N, wHHh, 4 * HH_N, Cb, 4 * HH_N);
  __syncthreads();

  // ---- hyper LSTM cell (elementwise) ----
  for (int i = tid; i < TM * HH_N; i += NTHR) {
    int r = i / HH_N, j = i % HH_N;
    size_t grow = (size_t)(row0 + r);
    float gi = Cb[r * 512 + j]            + hb[j];
    float gf = Cb[r * 512 + HH_N + j]     + hb[HH_N + j];
    float gg = Cb[r * 512 + 2 * HH_N + j] + hb[2 * HH_N + j];
    float go = Cb[r * 512 + 3 * HH_N + j] + hb[3 * HH_N + j];
    float nhc = sigf(gf) * hyper_c[grow * HH_N + j] + sigf(gi) * tanhf(gg);
    float nhh = sigf(go) * tanhf(nhc);
    o[OFF_HC + grow * HH_N + j] = nhc;
    o[OFF_HH + grow * HH_N + j] = nhh;
    hhb[i] = (_Float16)nhh;
  }
  __syncthreads();

  // ---- z = new_hyper_h @ projW^T + proj_b  (projW viewed as 384 x 128) ----
  gemm_tile<HH_N, false>(hhb, HH_N, wProj, 384, Cb, 384);
  __syncthreads();
  for (int i = tid; i < TM * 384; i += NTHR)
    zb[i] = (_Float16)(Cb[i] + proj_b[i % 384]);
  __syncthreads();

  // ---- gates: g in {i,f,g,o}: xg*hv_g + hgm*hv_{4+g} + bias*hv_{8+g}, LN, act
  for (int g = 0; g < 4; ++g) {
    gemm_tile<H_N, false>(xh, 512, wIh + (size_t)g * H_N * H_N, H_N, gb, H_N);
    gemm_tile<E_N, false>(zb + g * E_N, 384,
                          wScale + (size_t)g * H_N * E_N, H_N, hA, H_N);
    __syncthreads();
    for (int i = tid; i < TM * H_N; i += NTHR) gb[i] *= hA[i];
    __syncthreads();
    gemm_tile<H_N, false>(xh + 256, 512, wHh + (size_t)g * H_N * H_N, H_N, hB, H_N);
    gemm_tile<E_N, false>(zb + (4 + g) * E_N, 384,
                          wScale + (size_t)(4 + g) * H_N * E_N, H_N, hA, H_N);
    __syncthreads();
    for (int i = tid; i < TM * H_N; i += NTHR) gb[i] += hB[i] * hA[i];
    __syncthreads();
    gemm_tile<E_N, false>(zb + (8 + g) * E_N, 384,
                          wScale + (size_t)(8 + g) * H_N * E_N, H_N, hA, H_N);
    __syncthreads();
    for (int i = tid; i < TM * H_N; i += NTHR) {
      float v = gb[i] + bias[g * H_N + (i % H_N)] * hA[i];
      if (g == 1) v += 1.f;                 // forget-gate bias
      gb[i] = v;
    }
    __syncthreads();
    for (int r = wave; r < TM; r += 8) {    // wave-per-row LayerNorm + act
      float mean, istd;
      row_stats(gb + r * H_N, mean, istd);
#pragma unroll
      for (int q = 0; q < 8; ++q) {
        int d = lane + (q << 5);
        float v = (gb[r * H_N + d] - mean) * istd * gamma_ifgo[g * H_N + d]
                  + beta_ifgo[g * H_N + d];
        float a = (g == 2) ? tanhf(v) : sigf(v);
        act[((size_t)g * TM + r) * H_N + d] = (_Float16)a;
      }
    }
    __syncthreads();
  }

  // ---- LSTM cell: new_c = c*sig(f) + tanh(g)*sig(i); LN; new_h = tanh*sig(o)
  for (int i = tid; i < TM * H_N; i += NTHR) {
    int r = i / H_N, d = i % H_N;
    float iv = (float)act[((size_t)0 * TM + r) * H_N + d];
    float fv = (float)act[((size_t)1 * TM + r) * H_N + d];
    float gv = (float)act[((size_t)2 * TM + r) * H_N + d];
    gb[i] = c[(size_t)(row0 + r) * H_N + d] * fv + gv * iv;
  }
  __syncthreads();
  for (int r = wave; r < TM; r += 8) {
    float mean, istd;
    row_stats(gb + r * H_N, mean, istd);
    size_t grow = (size_t)(row0 + r);
#pragma unroll
    for (int q = 0; q < 8; ++q) {
      int d = lane + (q << 5);
      float ncn = (gb[r * H_N + d] - mean) * istd * gamma_c[d] + beta_c[d];
      float ov  = (float)act[((size_t)3 * TM + r) * H_N + d];
      float nh  = tanhf(ncn) * ov;
      o[OFF_NC + grow * H_N + d] = ncn;
      o[OFF_NH + grow * H_N + d] = nh;
      t1[r * H_N + d] = (_Float16)nh;
    }
  }
  __syncthreads();

  // ---- output = new_h @ out_W^T + out_b ----
  gemm_tile<H_N, false>(t1, H_N, wOut, NC_N, Cb, NC_N);
  __syncthreads();
  for (int i = tid; i < TM * NC_N; i += NTHR)
    o[OFF_OUT + (size_t)(row0 + i / NC_N) * NC_N + (i % NC_N)] =
        Cb[i] + out_b[i % NC_N];
}

// --------------------------------- launcher -----------------------------------

extern "C" void kernel_launch(void* const* d_in, const int* in_sizes, int n_in,
                              void* d_out, int out_size, void* d_ws, size_t ws_size,
                              hipStream_t stream) {
  (void)in_sizes; (void)n_in; (void)out_size; (void)ws_size;
  const float* x        = (const float*)d_in[0];
  const float* h        = (const float*)d_in[1];
  const float* c        = (const float*)d_in[2];
  const float* hyper_h  = (const float*)d_in[3];
  const float* hyper_c  = (const float*)d_in[4];
  const float* enc_W1   = (const float*)d_in[5];
  const float* enc_b1   = (const float*)d_in[6];
  const float* enc_W2   = (const float*)d_in[7];
  const float* enc_b2   = (const float*)d_in[8];
  const float* enc_W3   = (const float*)d_in[9];
  const float* enc_b3   = (const float*)d_in[10];
  const float* W_ih     = (const float*)d_in[11];
  const float* W_hh     = (const float*)d_in[12];
  const float* bias     = (const float*)d_in[13];
  const float* proj_W   = (const float*)d_in[14];
  const float* proj_b   = (const float*)d_in[15];
  const float* scale_W  = (const float*)d_in[16];
  const float* gamma_ifgo = (const float*)d_in[17];
  const float* beta_ifgo  = (const float*)d_in[18];
  const float* gamma_c  = (const float*)d_in[19];
  const float* beta_c   = (const float*)d_in[20];
  // d_in[21], d_in[22]: trans_W / trans_b — unused by the reference
  const float* out_W    = (const float*)d_in[23];
  const float* out_b    = (const float*)d_in[24];
  const float* hW_ih    = (const float*)d_in[25];
  const float* hW_hh    = (const float*)d_in[26];
  const float* hb       = (const float*)d_in[27];
  float* out = (float*)d_out;

  // f16 weight arena in workspace (~2.6 MB)
  _Float16* wsH = (_Float16*)d_ws;
  size_t off = 0;
  auto take = [&](size_t n) { _Float16* p = wsH + off; off += n; return p; };
  _Float16* wEnc1  = take((size_t)HL_N * OBS_N);
  _Float16* wEnc2  = take((size_t)HL_N * HL_N);
  _Float16* wEnc3  = take((size_t)HL_N * HL_N);
  _Float16* wIh    = take((size_t)4 * H_N * HL_N);
  _Float16* wHh    = take((size_t)4 * H_N * H_N);
  _Float16* wProj  = take((size_t)12 * E_N * HH_N);
  _Float16* wScale = take((size_t)12 * H_N * E_N);
  _Float16* wOut   = take((size_t)NC_N * H_N);
  _Float16* wHIh   = take((size_t)4 * HH_N * (HL_N + H_N));
  _Float16* wHHh   = take((size_t)4 * HH_N * HH_N);

  struct Cv { const float* s; _Float16* d; int n; };
  const Cv cv[10] = {
    { enc_W1,  wEnc1,  HL_N * OBS_N },
    { enc_W2,  wEnc2,  HL_N * HL_N },
    { enc_W3,  wEnc3,  HL_N * HL_N },
    { W_ih,    wIh,    4 * H_N * HL_N },
    { W_hh,    wHh,    4 * H_N * H_N },
    { proj_W,  wProj,  12 * E_N * HH_N },
    { scale_W, wScale, 12 * H_N * E_N },
    { out_W,   wOut,   NC_N * H_N },
    { hW_ih,   wHIh,   4 * HH_N * (HL_N + H_N) },
    { hW_hh,   wHHh,   4 * HH_N * HH_N },
  };
  for (int i = 0; i < 10; ++i) {
    int blocks = (cv[i].n + 255) / 256;
    f32_to_f16_kernel<<<blocks, 256, 0, stream>>>(cv[i].s, cv[i].d, cv[i].n);
  }

  constexpr int SMEM_BYTES =
      TM * (512 + 128 + 256 + 256 + 128 + 384 + 4 * 256) * 2  // f16 buffers
      + TM * (512 + 256) * 4;                                 // f32 buffers
  (void)hipFuncSetAttribute((const void*)hyperlstm_fused,
                            hipFuncAttributeMaxDynamicSharedMemorySize,
                            SMEM_BYTES);

  hyperlstm_fused<<<B_N / TM, NTHR, SMEM_BYTES, stream>>>(
      x, h, c, hyper_h, hyper_c,
      enc_b1, enc_b2, enc_b3,
      bias, proj_b, gamma_ifgo, beta_ifgo, gamma_c, beta_c, out_b, hb,
      wEnc1, wEnc2, wEnc3, wIh, wHh, wProj, wScale, wOut, wHIh, wHHh,
      out);
}
